// Yolov2_60301340836352
// MI455X (gfx1250) — compile-verified, matrix operand order checked by reading, not run on hardware
//
#include <hip/hip_runtime.h>
#include <hip/hip_bf16.h>

typedef _Float16 half_t;
typedef __attribute__((ext_vector_type(16))) _Float16 v16h;
typedef __attribute__((ext_vector_type(8)))  _Float16 v8h;
typedef __attribute__((ext_vector_type(8)))  float    v8f;
typedef __attribute__((ext_vector_type(4)))  int      i32x4;

// typed address-space pointers for the async-LDS builtin (param type per hipcc:
// '__attribute__((__vector_size__(4 * sizeof(int)))) int __device__ *')
typedef __attribute__((address_space(1))) i32x4* as1_i32x4_p;
typedef __attribute__((address_space(3))) i32x4* as3_i32x4_p;
typedef __attribute__((address_space(1))) void*  as1_void_p;
typedef __attribute__((address_space(3))) void*  as3_void_p;

#define TILE_M 64
#define TILE_N 128
#define TILE_K 32
#define LDS_KP 40   // padded K pitch (halves): 80B = 20 banks -> conflict-free frag loads

#if defined(__has_builtin)
#  if __has_builtin(__builtin_amdgcn_global_load_async_to_lds_b128)
#    define HAS_ASYNC_LDS 1
#  endif
#endif
#ifndef HAS_ASYNC_LDS
#  define HAS_ASYNC_LDS 0
#endif

static inline int ceil_div_h(int a, int b){ return (a + b - 1)/b; }

__device__ inline void wait_async_lds(){
#if defined(__has_builtin)
#  if __has_builtin(__builtin_amdgcn_s_wait_asynccnt)
  __builtin_amdgcn_s_wait_asynccnt(0);
  return;
#  endif
#endif
  asm volatile("s_wait_asynccnt 0" ::: "memory");
}

// ---------------------------------------------------------------------------
// prep / elementwise kernels
// ---------------------------------------------------------------------------
__global__ void k_cvt_f32_to_f16(const float* __restrict__ in, half_t* __restrict__ out, long n){
  long i  = (long)blockIdx.x*blockDim.x + threadIdx.x;
  long st = (long)gridDim.x*blockDim.x;
  for (; i < n; i += st) out[i] = (half_t)in[i];
}

// scale = g * rsqrt(v+eps); bias = (conv_bias - m)*scale + b   (conv_bias optional)
__global__ void k_bn_fold(const float* __restrict__ g, const float* __restrict__ b,
                          const float* __restrict__ m, const float* __restrict__ v,
                          const float* __restrict__ cbias,
                          float* __restrict__ scale, float* __restrict__ bias, int C){
  int i = blockIdx.x*blockDim.x + threadIdx.x;
  if (i < C){
    float s  = g[i]*rsqrtf(v[i] + 1e-5f);
    scale[i] = s;
    float cb = cbias ? cbias[i] : 0.0f;
    bias[i]  = (cb - m[i])*s + b[i];
  }
}

// 3x3 stride-2 pad-1 max pool on f16 NCHW
__global__ void k_maxpool3s2(const half_t* __restrict__ in, half_t* __restrict__ out,
                             int BC, int H, int W, int Ho, int Wo){
  int i   = blockIdx.x*blockDim.x + threadIdx.x;
  int tot = BC*Ho*Wo;
  if (i >= tot) return;
  int ox = i % Wo; int t = i / Wo; int oy = t % Ho; int bc = t / Ho;
  float mx = -3.4e38f;
  const half_t* base = in + (size_t)bc*H*W;
  #pragma unroll
  for (int dy = 0; dy < 3; ++dy){
    int iy = oy*2 - 1 + dy;
    if (iy < 0 || iy >= H) continue;
    #pragma unroll
    for (int dx = 0; dx < 3; ++dx){
      int ix = ox*2 - 1 + dx;
      if (ix < 0 || ix >= W) continue;
      mx = fmaxf(mx, (float)base[(size_t)iy*W + ix]);
    }
  }
  out[i] = (half_t)mx;
}

__constant__ float c_anchors[5][2] = {
  {1.3221f, 1.73145f}, {3.19275f, 4.00944f}, {5.05587f, 8.09892f},
  {9.47112f, 4.84053f}, {11.2364f, 10.0071f}};

// The reference's reshape of the NCHW head output is a raw flat reinterpret,
// so output flat index == input flat index; decode per channel slot.
__global__ void k_decode(const float* __restrict__ p, float* __restrict__ out){
  int i = blockIdx.x*blockDim.x + threadIdx.x;
  const int TOT = 4*13*13*5*25;
  if (i >= TOT) return;
  float t  = p[i];
  int  c   = i % 25;
  int  a   = (i/25) % 5;
  int  gx  = (i/125) % 13;         // second S axis ("cols")
  int  gy  = (i/(125*13)) % 13;    // first S axis ("rows")
  float r;
  if (c == 0)      r = 1.f/(1.f + __expf(-t)) + (float)gy;
  else if (c == 1) r = 1.f/(1.f + __expf(-t)) + (float)gx;
  else if (c == 2) r = __expf(t) * c_anchors[a][0];
  else if (c == 3) r = __expf(t) * c_anchors[a][1];
  else if (c == 4) r = 1.f/(1.f + __expf(-t));
  else             r = t;
  out[i] = r;
}

// ---------------------------------------------------------------------------
// implicit-GEMM convolution with v_wmma_f32_16x16x32_f16
//   GEMM: C[M=Cout][N=B*Ho*Wo] = W[M][K=Cin*KSZ*KSZ] * im2col[K][N]
//   Workgroup tile 64x128, 8 waves each owning a 32x32 C block:
//     2 A-frags x 2 B-frags -> 4 WMMAs per K-step.
//   Double-buffered LDS; weight tile staged with GLOBAL_LOAD_ASYNC_TO_LDS_B128
//   (ASYNCcnt) when K is 16B-granular, overlapping global latency with WMMA.
//   Fragment layout per CDNA5 ISA 7.12.2:
//     16-bit A frag (v16h): lane<16 -> K {0..7,16..23}, lane>=16 -> K {8..15,24..31}
//     (B mirrored with N in place of M); f32 C/D: VGPR r -> M = r (+8 for lane>=16)
// ---------------------------------------------------------------------------
__device__ inline v16h frag_from_lds(const half_t* rowbase, int kb2){
  v8h lo = *reinterpret_cast<const v8h*>(rowbase + kb2);
  v8h hi = *reinterpret_cast<const v8h*>(rowbase + kb2 + 16);
  v16h r;
  #pragma unroll
  for (int e = 0; e < 8; ++e){ r[e] = lo[e]; r[e+8] = hi[e]; }
  return r;
}

// ACT: 0=none 1=relu 2=leaky(0.1); RES: add residual; OF32: write fp32; KSZ: 1/3/7
template <int ACT, bool RES, bool OF32, int KSZ>
__global__ __launch_bounds__(256)
void k_conv_wmma(const half_t* __restrict__ in, const half_t* __restrict__ wgt,
                 const float* __restrict__ scale, const float* __restrict__ bias,
                 const half_t* __restrict__ res,
                 half_t* __restrict__ outh, float* __restrict__ outf,
                 int Bn, int Cin, int Hin, int Win,
                 int Cout, int Hout, int Wout,
                 int stride, int pad)
{
  __shared__ __align__(16) half_t As[2][TILE_M][LDS_KP];
  __shared__ __align__(16) half_t Bs[2][TILE_N][LDS_KP];

  constexpr int KHW = KSZ*KSZ;
  const int K    = Cin*KHW;
  const int HWi  = Hin*Win;
  const int HWo  = Hout*Wout;
  const int N    = Bn*HWo;
  const int m0   = blockIdx.y * TILE_M;
  const int n0   = blockIdx.x * TILE_N;
  const int tid  = threadIdx.x;
  const int lane = tid & 31;
  const int wave = tid >> 5;             // 8 waves
  const int wm   = (wave & 1) * 32;      // wave's M offset (2 groups x 32)
  const int wn   = (wave >> 1) * 32;     // wave's N offset (4 groups x 32)

  v8f acc[2][2] = {{{}, {}}, {{}, {}}};

  // A tile loader mapping: 64 rows x 32 k, one 16B chunk (8 halves) / thread
  const int arow = tid >> 2;             // 0..63
  const int akk  = (tid & 3) * 8;        // 0,8,16,24
  const bool kAligned = ((K & 7) == 0);  // uniform: async 16B chunks legal
  // B tile loader mapping: 128 rows x 32 k, 16 halves/thread
  const int brow = tid >> 1;             // 0..127
  const int bkk  = (tid & 1) * 16;       // 0,16

  // decompose this thread's B-column (output pixel) once
  const int nB = n0 + brow;
  int pb = 0, py = 0, px = 0; const bool nok = nB < N;
  if (nok){ pb = nB / HWo; int r = nB - pb*HWo; py = r / Wout; px = r - py*Wout; }
  size_t base1x1 = 0;
  if (KSZ == 1 && nok)   // 1x1 convs are pad=0 and always in-range
    base1x1 = (size_t)pb*Cin*HWi + (size_t)(py*stride)*Win + (size_t)(px*stride);

  const int kTiles = (K + TILE_K - 1)/TILE_K;

  auto load_tiles = [&](int kt, int buf){
    const int kb = kt*TILE_K;
    {   // ---- A tile (weights, row-major [Cout][K]) ----
      const int m = m0 + arow;
      half_t* dst = &As[buf][arow][akk];
#if HAS_ASYNC_LDS
      if (kAligned){
        if (m < Cout && (kb + akk) < K){
          __builtin_amdgcn_global_load_async_to_lds_b128(
              (as1_i32x4_p)(as1_void_p)(wgt + (size_t)m*K + kb + akk),
              (as3_i32x4_p)(as3_void_p)dst, 0, 0);
        } else {
          v8h z = {};
          *reinterpret_cast<v8h*>(dst) = z;
        }
      } else
#endif
      {
        const bool mok = m < Cout;
        #pragma unroll
        for (int u = 0; u < 8; ++u){
          const int k = kb + akk + u;
          dst[u] = (mok && k < K) ? wgt[(size_t)m*K + k] : (half_t)0.0f;
        }
        if (kt + 1 < kTiles && mok)
          __builtin_prefetch(&wgt[(size_t)m*K + kb + TILE_K + akk], 0, 3);
      }
    }
    {   // ---- B tile (im2col gather, stored [N][K] = K-contiguous) ----
      if (KSZ == 1){
        #pragma unroll
        for (int u = 0; u < 16; ++u){
          const int k = kb + bkk + u;
          Bs[buf][brow][bkk+u] = (nok && k < K) ? in[base1x1 + (size_t)k*HWi] : (half_t)0.0f;
        }
      } else {
        #pragma unroll
        for (int u = 0; u < 16; ++u){
          const int k = kb + bkk + u;
          half_t vv = (half_t)0.0f;
          if (nok && k < K){
            const int ci = k / KHW;              // compile-time divisor
            const int r2 = k - ci*KHW;
            const int kh = r2 / KSZ;             // compile-time divisor
            const int kw = r2 - kh*KSZ;
            const int iy = py*stride - pad + kh;
            const int ix = px*stride - pad + kw;
            if (iy >= 0 && iy < Hin && ix >= 0 && ix < Win)
              vv = in[(size_t)pb*Cin*HWi + (size_t)ci*HWi + (size_t)iy*Win + ix];
          }
          Bs[buf][brow][bkk+u] = vv;
        }
      }
    }
  };

  // prologue: stage tile 0
  load_tiles(0, 0);
  wait_async_lds();
  __syncthreads();

  const int kb2 = (lane & 16) ? 8 : 0;
  const int l15 = lane & 15;

  for (int kt = 0; kt < kTiles; ++kt){
    const int buf = kt & 1;
    v16h af[2], bf[2];
    af[0] = frag_from_lds(&As[buf][wm      + l15][0], kb2);
    af[1] = frag_from_lds(&As[buf][wm + 16 + l15][0], kb2);
    bf[0] = frag_from_lds(&Bs[buf][wn      + l15][0], kb2);
    bf[1] = frag_from_lds(&Bs[buf][wn + 16 + l15][0], kb2);

    if (kt + 1 < kTiles) load_tiles(kt + 1, buf ^ 1);  // overlap with WMMA

    #pragma unroll
    for (int ai = 0; ai < 2; ++ai)
      #pragma unroll
      for (int bi = 0; bi < 2; ++bi)
        acc[ai][bi] = __builtin_amdgcn_wmma_f32_16x16x32_f16(
            false, af[ai], false, bf[bi], (short)0, acc[ai][bi], false, false);

    wait_async_lds();
    __syncthreads();
  }

  // ---- fused epilogue: BN scale/bias (+conv bias), residual, activation ----
  const int mhalf = (lane & 16) ? 8 : 0;
  float scv[2][8], biv[2][8];
  #pragma unroll
  for (int ai = 0; ai < 2; ++ai)
    #pragma unroll
    for (int r = 0; r < 8; ++r){
      const int m = m0 + wm + 16*ai + mhalf + r;
      const bool mok = m < Cout;
      scv[ai][r] = mok ? scale[m] : 0.0f;
      biv[ai][r] = mok ? bias[m]  : 0.0f;
    }
  #pragma unroll
  for (int bj = 0; bj < 2; ++bj){
    const int n = n0 + wn + 16*bj + l15;
    if (n >= N) continue;
    const int    bb    = n / HWo;          // hoisted: 2 divisions, not 32
    const int    rem   = n - bb*HWo;
    const size_t obase = (size_t)bb*Cout*HWo + rem;
    #pragma unroll
    for (int ai = 0; ai < 2; ++ai){
      #pragma unroll
      for (int r = 0; r < 8; ++r){
        const int m = m0 + wm + 16*ai + mhalf + r;
        if (m >= Cout) continue;
        float vv = acc[ai][bj][r] * scv[ai][r] + biv[ai][r];
        const size_t o = obase + (size_t)m*HWo;
        if (RES)           vv += (float)res[o];
        if (ACT == 1)      vv = vv > 0.f ? vv : 0.f;
        else if (ACT == 2) vv = vv > 0.f ? vv : 0.1f*vv;
        if (OF32) outf[o] = vv;
        else      outh[o] = (half_t)vv;
      }
    }
  }
}

// ---------------------------------------------------------------------------
// host orchestration
// ---------------------------------------------------------------------------
extern "C" void kernel_launch(void* const* d_in, const int* in_sizes, int n_in,
                              void* d_out, int out_size, void* d_ws, size_t ws_size,
                              hipStream_t stream)
{
  (void)in_sizes; (void)n_in; (void)out_size; (void)ws_size;
  int idx = 0;
  auto nxt = [&]() -> const float* { return (const float*)d_in[idx++]; };

  const float* x = nxt();

  // workspace carve (halves): 3x max activation (11.08M), max weight (18.87M)
  uint8_t* wsp = (uint8_t*)d_ws;
  auto carve = [&](size_t bytes) -> void* {
    void* p = (void*)wsp;
    wsp += (bytes + 255) & ~(size_t)255;
    return p;
  };
  const size_t MAXACT = 11075584ull;   // 4*64*208*208 (== 4*256*104*104) halves
  const size_t MAXWEL = 18874368ull;   // 2048*1024*3*3 halves
  half_t* xh   = (half_t*)carve(2076672ull*2);
  half_t* bufA = (half_t*)carve(MAXACT*2);
  half_t* bufB = (half_t*)carve(MAXACT*2);
  half_t* bufC = (half_t*)carve(MAXACT*2);
  half_t* wst  = (half_t*)carve(MAXWEL*2);
  float*  ssc  = (float*)carve(2048*4);
  float*  sbi  = (float*)carve(2048*4);
  float*  fout = (float*)carve(84500ull*4);

  auto cvt = [&](const float* src, half_t* dst, long n){
    int blocks = (int)((n + 255)/256); if (blocks > 4096) blocks = 4096;
    k_cvt_f32_to_f16<<<blocks, 256, 0, stream>>>(src, dst, n);
  };

  auto run_conv = [&](const float* w, const float* g, const float* b, const float* m,
                      const float* v, const float* cbias,
                      const half_t* in, half_t* outh, float* outf, const half_t* res,
                      int Bn, int Cin, int H, int W, int Cout,
                      int KH, int stride, int pad,
                      int act, bool use_res, bool of32, int& Ho, int& Wo){
    Ho = (H + 2*pad - KH)/stride + 1;
    Wo = (W + 2*pad - KH)/stride + 1;
    cvt(w, wst, (long)Cout*Cin*KH*KH);
    k_bn_fold<<<ceil_div_h(Cout,256), 256, 0, stream>>>(g,b,m,v,cbias, ssc,sbi, Cout);
    int N = Bn*Ho*Wo;
    dim3 grid(ceil_div_h(N, TILE_N), ceil_div_h(Cout, TILE_M), 1);
    dim3 block(256,1,1);
#define LAUNCH_CONV(A_, R_, O_, KS_) \
    k_conv_wmma<A_,R_,O_,KS_><<<grid,block,0,stream>>>( \
        in,wst,ssc,sbi,res,outh,outf,Bn,Cin,H,W,Cout,Ho,Wo,stride,pad)
    if (KH == 7)                      LAUNCH_CONV(1,false,false,7);
    else if (KH == 3 && act == 1)     LAUNCH_CONV(1,false,false,3);
    else if (KH == 3)                 LAUNCH_CONV(2,false,false,3);
    else if (of32)                    LAUNCH_CONV(2,false,true ,1);
    else if (use_res)                 LAUNCH_CONV(1,true ,false,1);
    else if (act == 0)                LAUNCH_CONV(0,false,false,1);
    else if (act == 1)                LAUNCH_CONV(1,false,false,1);
    else                              LAUNCH_CONV(2,false,false,1);
#undef LAUNCH_CONV
  };

  // ---- stem: conv 7x7 s2 p3 + BN + ReLU, then maxpool 3x3 s2 p1 ----
  cvt(x, xh, 2076672L);
  const float* stem_w = nxt();
  const float* sg = nxt(); const float* sb = nxt(); const float* sm = nxt(); const float* sv = nxt();
  int Ho, Wo;
  run_conv(stem_w, sg, sb, sm, sv, nullptr, xh, bufB, nullptr, nullptr,
           4, 3, 416, 416, 64, 7, 2, 3, /*act*/1, false, false, Ho, Wo);  // bufB: 64x208x208
  k_maxpool3s2<<<ceil_div_h(4*64*104*104,256), 256, 0, stream>>>(bufB, bufA, 4*64, 208, 208, 104, 104);

  // ---- ResNet-50 bottleneck layers ----
  int H = 104, W = 104, in_c = 64;
  half_t* cur = bufA; half_t* t1 = bufB; half_t* t2 = bufC;
  const int mids[4]    = {64, 128, 256, 512};
  const int nblk[4]    = {3, 4, 6, 3};
  const int lstride[4] = {1, 2, 2, 2};
  for (int l = 0; l < 4; ++l){
    const int mid = mids[l];
    for (int j = 0; j < nblk[l]; ++j){
      const int s = (j == 0) ? lstride[l] : 1;
      const float* w1 = nxt(); const float* g1 = nxt(); const float* b1 = nxt(); const float* m1 = nxt(); const float* v1 = nxt();
      const float* w2 = nxt(); const float* g2 = nxt(); const float* b2 = nxt(); const float* m2 = nxt(); const float* v2 = nxt();
      const float* w3 = nxt(); const float* g3 = nxt(); const float* b3 = nxt(); const float* m3 = nxt(); const float* v3 = nxt();
      const float *wd = nullptr, *gd = nullptr, *bd = nullptr, *md = nullptr, *vd = nullptr;
      if (j == 0){ wd = nxt(); gd = nxt(); bd = nxt(); md = nxt(); vd = nxt(); }

      int h1, w1o, h2, w2o;
      run_conv(w1, g1, b1, m1, v1, nullptr, cur, t1, nullptr, nullptr,
               4, in_c, H, W, mid, 1, 1, 0, 1, false, false, h1, w1o);
      run_conv(w2, g2, b2, m2, v2, nullptr, t1, t2, nullptr, nullptr,
               4, mid, H, W, mid, 3, s, 1, 1, false, false, h2, w2o);
      const half_t* resp = cur;
      if (j == 0){
        int hd, wdo;
        run_conv(wd, gd, bd, md, vd, nullptr, cur, t1, nullptr, nullptr,
                 4, in_c, H, W, mid*4, 1, s, 0, 0, false, false, hd, wdo);
        resp = t1;
      }
      int h3, w3o;   // conv3: BN + residual + ReLU, written in place over `cur`
      run_conv(w3, g3, b3, m3, v3, nullptr, t2, cur, nullptr, resp,
               4, mid, h2, w2o, mid*4, 1, 1, 0, 1, true, false, h3, w3o);
      in_c = mid*4; H = h2; W = w2o;
    }
  }

  // ---- detection head: 5x (conv+bias -> BN -> LeakyReLU) + final 1x1 ----
  const int hic[6] = {2048, 1024, 2048, 1024, 2048, 1024};
  const int hoc[6] = {1024, 2048, 1024, 2048, 1024, 125};
  const int hk [6] = {3, 3, 3, 3, 3, 1};
  const int hp [6] = {1, 1, 1, 1, 1, 0};
  half_t* hin = cur; half_t* hout = t1;
  for (int i = 0; i < 6; ++i){
    const float* w  = nxt(); const float* cb = nxt();
    const float* g  = nxt(); const float* b  = nxt(); const float* m = nxt(); const float* v = nxt();
    int ho, wo;
    if (i < 5){
      run_conv(w, g, b, m, v, cb, hin, hout, nullptr, nullptr,
               4, hic[i], 13, 13, hoc[i], hk[i], 1, hp[i], 2, false, false, ho, wo);
      half_t* t = hin; hin = hout; hout = t;
    } else {
      run_conv(w, g, b, m, v, cb, hin, nullptr, fout, nullptr,
               4, hic[i], 13, 13, hoc[i], hk[i], 1, hp[i], 2, false, true, ho, wo);
    }
  }

  // ---- decode ----
  k_decode<<<ceil_div_h(4*13*13*125, 256), 256, 0, stream>>>(fout, (float*)d_out);
}
